// MambaBlock_50861002719354
// MI455X (gfx1250) — matrix operand order repaired and checked by reading, not compiled, not run on hardware
//
#include <hip/hip_runtime.h>
#include <hip/hip_bf16.h>

#define D_MODEL 1024
#define D_STATE 16
#define D_CONV  4
#define D_INNER 2048
#define BATCH   2
#define SEQ     2048
#define NTOK    (BATCH * SEQ)   /* 4096 tokens */

typedef __bf16 bf16_t;
typedef __attribute__((ext_vector_type(16))) __bf16 v16bf;
typedef __attribute__((ext_vector_type(8)))  __bf16 v8bf;
typedef __attribute__((ext_vector_type(8)))  float  v8f;

// ---------------- workspace layout (bytes, 256-aligned) ----------------
static constexpr size_t O_XN   = 0;                                          // bf16 4096x1024
static constexpr size_t O_WIN  = O_XN   + (size_t)NTOK * D_MODEL * 2;        // bf16 4096x1024
static constexpr size_t O_WX   = O_WIN  + (size_t)2 * D_INNER * D_MODEL * 2; // bf16 32x2048
static constexpr size_t O_WDT  = O_WX   + 256 * 1024;                        // bf16 2048x2048
static constexpr size_t O_WOUT = O_WDT  + (size_t)D_INNER * D_INNER * 2;     // bf16 1024x2048
static constexpr size_t O_XZ   = O_WOUT + (size_t)D_MODEL * D_INNER * 2;     // f32  4096x4096
static constexpr size_t O_XCF  = O_XZ   + (size_t)NTOK * 2 * D_INNER * 4;    // f32  4096x2048
static constexpr size_t O_XCB  = O_XCF  + (size_t)NTOK * D_INNER * 4;        // bf16 4096x2048
static constexpr size_t O_DT   = O_XCB  + (size_t)NTOK * D_INNER * 2;        // f32  4096x2048
static constexpr size_t O_XDBL = O_DT   + (size_t)NTOK * D_INNER * 4;        // f32  4096x32
static constexpr size_t O_Y    = O_XDBL + 1024 * 1024;                       // f32  4096x2048
static constexpr size_t O_YG   = O_Y    + (size_t)NTOK * D_INNER * 4;        // bf16 4096x2048

// ---------------- f32 -> bf16 convert ----------------
__global__ __launch_bounds__(256)
void cvt_bf16_kernel(const float* __restrict__ in, bf16_t* __restrict__ out, int n) {
    int i = blockIdx.x * 256 + threadIdx.x;
    if (i < n) out[i] = (bf16_t)in[i];
}

// ---------------- LayerNorm -> bf16 ----------------
__global__ __launch_bounds__(256)
void layernorm_bf16_kernel(const float* __restrict__ x, const float* __restrict__ g,
                           const float* __restrict__ b, bf16_t* __restrict__ out) {
    __shared__ float s1[256], s2[256];
    const int t = blockIdx.x;
    const float* row = x + (size_t)t * D_MODEL;
    float v[4], sum = 0.f, sq = 0.f;
#pragma unroll
    for (int i = 0; i < 4; ++i) {
        v[i] = row[threadIdx.x + i * 256];
        sum += v[i]; sq += v[i] * v[i];
    }
    s1[threadIdx.x] = sum; s2[threadIdx.x] = sq;
    __syncthreads();
    for (int off = 128; off > 0; off >>= 1) {
        if ((int)threadIdx.x < off) {
            s1[threadIdx.x] += s1[threadIdx.x + off];
            s2[threadIdx.x] += s2[threadIdx.x + off];
        }
        __syncthreads();
    }
    const float mu   = s1[0] * (1.0f / D_MODEL);
    const float var  = s2[0] * (1.0f / D_MODEL) - mu * mu;
    const float rstd = rsqrtf(var + 1e-5f);
    bf16_t* orow = out + (size_t)t * D_MODEL;
#pragma unroll
    for (int i = 0; i < 4; ++i) {
        int c = threadIdx.x + i * 256;
        orow[c] = (bf16_t)((v[i] - mu) * rstd * g[c] + b[c]);
    }
}

// A-fragment load: lane<16 holds row M=lr, K {0-7,16-23}; lane>=16 K {8-15,24-31}
__device__ __forceinline__ v16bf load_frag_a(const bf16_t* p) {
    v8bf a0 = *(const v8bf*)(p);
    v8bf a1 = *(const v8bf*)(p + 16);
    v16bf r;
#pragma unroll
    for (int i = 0; i < 8; ++i) { r[i] = a0[i]; r[8 + i] = a1[i]; }
    return r;
}
// B-fragment load: lane holds one column (= one contiguous NxK weight row),
// lane<16 -> K {0-7,16-23}, lane>=16 -> K {8-15,24-31}
__device__ __forceinline__ v16bf load_frag_b(const bf16_t* p) {
    v8bf b0 = *(const v8bf*)(p);
    v8bf b1 = *(const v8bf*)(p + 16);
    v16bf r;
#pragma unroll
    for (int i = 0; i < 8; ++i) { r[i] = b0[i]; r[8 + i] = b1[i]; }
    return r;
}

// ================= register-blocked bf16 WMMA GEMM =================
// One wave computes a 64x64 output block = 4x4 tiles of 16x16 (128 accumulator
// VGPRs). Per K-step of 32: 8 fragment loads feed 16 v_wmma ops (1 load : 2 WMMA).
// Epilogue is compile-time specialized (<BIAS, RES, ACT>) to keep register
// pressure below the 256-VGPR direct window (no accumulator spill).
// C[M,N] = A[M,K] * W[N,K]^T. M,N % 64 == 0, K % 32 == 0.
template <bool BIAS, bool RES, int ACT>
__global__ __launch_bounds__(32)
void wmma_gemm_bf16_blk_kernel(const bf16_t* __restrict__ A, const bf16_t* __restrict__ W,
                               float* __restrict__ C, int M, int N, int K,
                               const float* __restrict__ bias,
                               const float* __restrict__ residual) {
    const int tn   = blockIdx.x << 6;
    const int tm   = blockIdx.y << 6;
    const int lane = threadIdx.x & 31;
    const int half = lane >> 4;
    const int lr   = lane & 15;

    const bf16_t* arow[4];
    const bf16_t* brow[4];
#pragma unroll
    for (int i = 0; i < 4; ++i)
        arow[i] = A + (size_t)(tm + (i << 4) + lr) * K + (half << 3);
#pragma unroll
    for (int j = 0; j < 4; ++j)
        brow[j] = W + (size_t)(tn + (j << 4) + lr) * K + (half << 3);

    v8f c[4][4] = {};
    for (int k0 = 0; k0 < K; k0 += 32) {
        v16bf av[4], bv[4];
#pragma unroll
        for (int i = 0; i < 4; ++i) av[i] = load_frag_a(arow[i] + k0);
#pragma unroll
        for (int j = 0; j < 4; ++j) bv[j] = load_frag_b(brow[j] + k0);
#pragma unroll
        for (int i = 0; i < 4; ++i)
#pragma unroll
            for (int j = 0; j < 4; ++j)
                c[i][j] = __builtin_amdgcn_wmma_f32_16x16x32_bf16(
                    false, av[i], false, bv[j], (short)0, c[i][j], false, false);
    }

    // ---- epilogue: one row base per output row, j innermost (stride 64B) ----
    float bvals[4];
#pragma unroll
    for (int j = 0; j < 4; ++j) bvals[j] = BIAS ? bias[tn + (j << 4) + lr] : 0.0f;

#pragma unroll
    for (int i = 0; i < 4; ++i) {
#pragma unroll
        for (int r = 0; r < 8; ++r) {
            const int row = tm + (i << 4) + r + (half << 3);
            float* crow = C + (size_t)row * N + tn + lr;
            const float* rrow = RES ? (residual + (size_t)row * N + tn + lr) : nullptr;
#pragma unroll
            for (int j = 0; j < 4; ++j) {
                float v = c[i][j][r];
                if (BIAS) v += bvals[j];
                if (ACT == 1) v = (v > 20.f) ? v : log1pf(__expf(v));   // softplus
                if (RES)  v += rrow[j << 4];
                crow[j << 4] = v;
            }
        }
    }
}

// ---------------- simple 16x16-per-wave WMMA GEMM (for skinny N=32 case) ----------------
__global__ __launch_bounds__(32)
void wmma_gemm_bf16_kernel(const bf16_t* __restrict__ A, const bf16_t* __restrict__ W,
                           float* __restrict__ C, int M, int N, int K) {
    const int tn   = blockIdx.x << 4;
    const int tm   = blockIdx.y << 4;
    const int lane = threadIdx.x & 31;
    const int half = lane >> 4;
    const int lr   = lane & 15;

    const bf16_t* arow = A + (size_t)(tm + lr) * K + (half << 3);
    const bf16_t* brow = W + (size_t)(tn + lr) * K + (half << 3);

    v8f c = {};
    for (int k0 = 0; k0 < K; k0 += 32) {
        v16bf av = load_frag_a(arow + k0);
        v16bf bv = load_frag_b(brow + k0);
        c = __builtin_amdgcn_wmma_f32_16x16x32_bf16(false, av, false, bv,
                                                    (short)0, c, false, false);
    }
    const int col = tn + lr;
#pragma unroll
    for (int r = 0; r < 8; ++r) {
        const int row = tm + r + (half << 3);
        C[(size_t)row * N + col] = c[r];
    }
}

// ---------------- causal depthwise conv (k=4) + SiLU -> f32 + bf16 ----------------
__global__ __launch_bounds__(256)
void conv_silu_kernel(const float* __restrict__ xz, const float* __restrict__ w,
                      const float* __restrict__ cb, float* __restrict__ of,
                      bf16_t* __restrict__ ob) {
    const size_t idx = (size_t)blockIdx.x * 256 + threadIdx.x;   // over NTOK*D_INNER
    const int d = (int)(idx & (D_INNER - 1));
    const int t = (int)(idx >> 11);
    const int l = t & (SEQ - 1);
    float acc = cb[d];
#pragma unroll
    for (int j = 0; j < D_CONV; ++j) {
        const int ls = l - (D_CONV - 1) + j;
        if (ls >= 0)
            acc += w[d * D_CONV + j] * xz[(size_t)(t - l + ls) * (2 * D_INNER) + d];
    }
    const float sv = acc * (1.0f / (1.0f + __expf(-acc)));       // SiLU
    of[idx] = sv;
    ob[idx] = (bf16_t)sv;
}

// ---------------- selective scan: 4096 independent (b,d) rows, 16 states in regs ----------
__global__ __launch_bounds__(256)
void ssm_scan_kernel(const float* __restrict__ dt, const float* __restrict__ xc,
                     const float* __restrict__ xdbl, const float* __restrict__ A_log,
                     const float* __restrict__ Dp, float* __restrict__ y) {
    const int tid = blockIdx.x * 256 + threadIdx.x;   // 0..4095
    const int b = tid >> 11;
    const int d = tid & (D_INNER - 1);
    float A[D_STATE], h[D_STATE];
#pragma unroll
    for (int n = 0; n < D_STATE; ++n) {
        A[n] = -__expf(A_log[d * D_STATE + n]);
        h[n] = 0.0f;
    }
    const float Dv = Dp[d];
    for (int l = 0; l < SEQ; ++l) {
        const size_t i = ((size_t)b * SEQ + l) * D_INNER + d;
        const float dtv = dt[i];
        const float xv  = xc[i];
        const float dx  = dtv * xv;
        const float* bc = xdbl + ((size_t)b * SEQ + l) * (2 * D_STATE);
        float acc = 0.0f;
#pragma unroll
        for (int n = 0; n < D_STATE; ++n) {
            h[n] = __expf(dtv * A[n]) * h[n] + dx * bc[n];       // A_bar*h + dt*x*B
            acc += h[n] * bc[D_STATE + n];                       // h . C
        }
        y[i] = acc + Dv * xv;
    }
}

// ---------------- y * silu(z) -> bf16 ----------------
__global__ __launch_bounds__(256)
void gate_silu_kernel(const float* __restrict__ y, const float* __restrict__ xz,
                      bf16_t* __restrict__ out) {
    const size_t idx = (size_t)blockIdx.x * 256 + threadIdx.x;
    const int d = (int)(idx & (D_INNER - 1));
    const int t = (int)(idx >> 11);
    const float z = xz[(size_t)t * (2 * D_INNER) + D_INNER + d];
    const float g = z * (1.0f / (1.0f + __expf(-z)));
    out[idx] = (bf16_t)(y[idx] * g);
}

extern "C" void kernel_launch(void* const* d_in, const int* in_sizes, int n_in,
                              void* d_out, int out_size, void* d_ws, size_t ws_size,
                              hipStream_t stream) {
    (void)in_sizes; (void)n_in; (void)out_size; (void)ws_size;
    const float* x      = (const float*)d_in[0];
    const float* W_in   = (const float*)d_in[1];
    const float* conv_w = (const float*)d_in[2];
    const float* conv_b = (const float*)d_in[3];
    const float* W_x    = (const float*)d_in[4];
    const float* W_dt   = (const float*)d_in[5];
    const float* b_dt   = (const float*)d_in[6];
    const float* A_log  = (const float*)d_in[7];
    const float* Dp     = (const float*)d_in[8];
    const float* W_out  = (const float*)d_in[9];
    const float* ln_g   = (const float*)d_in[10];
    const float* ln_b   = (const float*)d_in[11];
    float* out = (float*)d_out;

    char* ws = (char*)d_ws;
    bf16_t* xn_bf   = (bf16_t*)(ws + O_XN);
    bf16_t* Win_bf  = (bf16_t*)(ws + O_WIN);
    bf16_t* Wx_bf   = (bf16_t*)(ws + O_WX);
    bf16_t* Wdt_bf  = (bf16_t*)(ws + O_WDT);
    bf16_t* Wout_bf = (bf16_t*)(ws + O_WOUT);
    float*  xz      = (float*)(ws + O_XZ);
    float*  xc_f    = (float*)(ws + O_XCF);
    bf16_t* xc_bf   = (bf16_t*)(ws + O_XCB);
    float*  dtbuf   = (float*)(ws + O_DT);
    float*  xdbl    = (float*)(ws + O_XDBL);
    float*  ybuf    = (float*)(ws + O_Y);
    bf16_t* yg_bf   = (bf16_t*)(ws + O_YG);

    // 1) weights -> bf16
    {
        int n;
        n = 2 * D_INNER * D_MODEL;
        cvt_bf16_kernel<<<(n + 255) / 256, 256, 0, stream>>>(W_in, Win_bf, n);
        n = 2 * D_STATE * D_INNER;
        cvt_bf16_kernel<<<(n + 255) / 256, 256, 0, stream>>>(W_x, Wx_bf, n);
        n = D_INNER * D_INNER;
        cvt_bf16_kernel<<<(n + 255) / 256, 256, 0, stream>>>(W_dt, Wdt_bf, n);
        n = D_MODEL * D_INNER;
        cvt_bf16_kernel<<<(n + 255) / 256, 256, 0, stream>>>(W_out, Wout_bf, n);
    }

    // 2) layernorm
    layernorm_bf16_kernel<<<NTOK, 256, 0, stream>>>(x, ln_g, ln_b, xn_bf);

    // 3) xz = xn @ W_in^T   (4096 x 4096 x 1024), 64x64 per wave
    wmma_gemm_bf16_blk_kernel<false, false, 0>
        <<<dim3((2 * D_INNER) / 64, NTOK / 64), 32, 0, stream>>>(
        xn_bf, Win_bf, xz, NTOK, 2 * D_INNER, D_MODEL, nullptr, nullptr);

    // 4) causal depthwise conv + SiLU
    conv_silu_kernel<<<(NTOK * D_INNER) / 256, 256, 0, stream>>>(
        xz, conv_w, conv_b, xc_f, xc_bf);

    // 5) x_dbl = x_branch @ W_x^T   (4096 x 32 x 2048), skinny -> simple kernel
    wmma_gemm_bf16_kernel<<<dim3((2 * D_STATE) / 16, NTOK / 16), 32, 0, stream>>>(
        xc_bf, Wx_bf, xdbl, NTOK, 2 * D_STATE, D_INNER);

    // 6) dt = softplus(x_branch @ W_dt^T + b_dt)   (4096 x 2048 x 2048)
    wmma_gemm_bf16_blk_kernel<true, false, 1>
        <<<dim3(D_INNER / 64, NTOK / 64), 32, 0, stream>>>(
        xc_bf, Wdt_bf, dtbuf, NTOK, D_INNER, D_INNER, b_dt, nullptr);

    // 7) selective scan
    ssm_scan_kernel<<<(BATCH * D_INNER) / 256, 256, 0, stream>>>(
        dtbuf, xc_f, xdbl, A_log, Dp, ybuf);

    // 8) gate: y * silu(z)
    gate_silu_kernel<<<(NTOK * D_INNER) / 256, 256, 0, stream>>>(ybuf, xz, yg_bf);

    // 9) out = yg @ W_out^T + residual   (4096 x 1024 x 2048)
    wmma_gemm_bf16_blk_kernel<false, true, 0>
        <<<dim3(D_MODEL / 64, NTOK / 64), 32, 0, stream>>>(
        yg_bf, Wout_bf, out, NTOK, D_MODEL, D_INNER, nullptr, x);
}
